// graph_rcnn_67774583931067
// MI455X (gfx1250) — matrix operand order repaired and checked by reading, not compiled, not run on hardware
//
#include <hip/hip_runtime.h>
#include <hip/hip_bf16.h>

// ---------------------------------------------------------------------------
// Graph R-CNN relation head on MI455X (gfx1250).
// Every large matmul runs through one tiled bf16 WMMA GEMM:
//   - v_wmma_f32_16x16x32_bf16, f32 accumulation
//   - B operand always pre-transposed ([N,K] row-major) -> uniform 128-bit
//     staging, no in-loop transposes
//   - double-buffered LDS tiles filled with GLOBAL_LOAD_ASYNC_TO_LDS_B128
//     (ASYNCcnt-tracked, s_wait_asynccnt) so HBM fetch overlaps the matrix pipe
// ---------------------------------------------------------------------------

typedef __attribute__((ext_vector_type(16))) __bf16 v16bf;
typedef __attribute__((ext_vector_type(8)))  __bf16 v8bf;
typedef __attribute__((ext_vector_type(8)))  float  v8f;

#define EPI_BF16       1
#define EPI_RELU_BF16  2
#define EPI_ADJ        3

// Async global->LDS 128-bit copy (per-lane), tracked by ASYNCcnt.
__device__ __forceinline__ void async_b128(unsigned lds_addr, const void* gptr) {
    asm volatile("global_load_async_to_lds_b128 %0, %1, off"
                 :: "v"(lds_addr), "v"(gptr)
                 : "memory");
}

// ---------------------------------------------------------------------------
// Tiled WMMA GEMM:  C[M,N] = epi(A[M,K] @ B^T + bias),  B given as [N,K].
//   block tile 64x64, 128 threads (4 waves), wave tile 32x32 (2x2 WMMA accums)
//   K stepped by 32 through double-buffered LDS, async-staged.
//   Batched via grid.z with element strides sAb/sBb/sCb.
//   Optional CgT: also store C transposed per 128-row group:
//     CgT[(row/128)*N*128 + col*128 + (row%128)]   (GNN node count = 128)
// Assumes M,N multiples of 64, K multiple of 32 (true for every call here).
// ---------------------------------------------------------------------------
template <int EPI>
__global__ __launch_bounds__(128)
void gemm_wmma_bf16(const __bf16* __restrict__ Ag, const __bf16* __restrict__ BTg,
                    const float* __restrict__ bias, __bf16* __restrict__ Cg,
                    __bf16* __restrict__ CgT,
                    int M, int N, int K,
                    long long sAb, long long sBb, long long sCb,
                    const int* __restrict__ num_obj)
{
    // [buf][row][k], 80B row stride (16B multiple) to dodge bank conflicts
    __shared__ __attribute__((aligned(16))) __bf16 sA[2][64][40];
    __shared__ __attribute__((aligned(16))) __bf16 sB[2][64][40];

    const int tid  = threadIdx.x;
    const int lane = tid & 31;
    const int wave = tid >> 5;
    const int gm   = blockIdx.y * 64;
    const int gn   = blockIdx.x * 64;
    const int bz   = blockIdx.z;

    const __bf16* A  = Ag  + (size_t)bz * sAb;
    const __bf16* BT = BTg + (size_t)bz * sBb;

    const int wm = (wave >> 1) * 32;   // wave sub-tile origin in block tile
    const int wn = (wave & 1)  * 32;

    // Per-thread staging chunks: 256 chunks of 8 bf16 per 64x32 tile.
    const int r0 = tid >> 2,           kc0 = (tid & 3) * 8;
    const int r1 = (tid + 128) >> 2,   kc1 = ((tid + 128) & 3) * 8;

    // Issue one stage's async copies (A + B tiles into buffer `buf`).
    auto issue_stage = [&](int k0, int buf) {
        async_b128((unsigned)(size_t)&sA[buf][r0][kc0],
                   A  + (size_t)(gm + r0) * K + (k0 + kc0));
        async_b128((unsigned)(size_t)&sA[buf][r1][kc1],
                   A  + (size_t)(gm + r1) * K + (k0 + kc1));
        async_b128((unsigned)(size_t)&sB[buf][r0][kc0],
                   BT + (size_t)(gn + r0) * K + (k0 + kc0));
        async_b128((unsigned)(size_t)&sB[buf][r1][kc1],
                   BT + (size_t)(gn + r1) * K + (k0 + kc1));
    };

    v8f acc[2][2] = {};
    const int nst = K >> 5;            // K / 32 stages

    issue_stage(0, 0);
    for (int s = 0; s < nst; ++s) {
        const int buf = s & 1;
        if (s + 1 < nst) {
            issue_stage((s + 1) << 5, buf ^ 1);
            asm volatile("s_wait_asynccnt 4" ::: "memory"); // stage s done, s+1 in flight
        } else {
            asm volatile("s_wait_asynccnt 0" ::: "memory");
        }
        __syncthreads();

        // A 16x32 bf16 frag: lane<16 -> row=lane,    K 0..7 then 16..23
        //                    lane>=16 -> row=lane-16, K 8..15 then 24..31
        v16bf afrag[2];
#pragma unroll
        for (int ti = 0; ti < 2; ++ti) {
            int row = wm + ti * 16 + (lane & 15);
            int kb  = (lane < 16) ? 0 : 8;
            v8bf lo = *(const v8bf*)&sA[buf][row][kb];
            v8bf hi = *(const v8bf*)&sA[buf][row][kb + 16];
            afrag[ti] = __builtin_shufflevector(lo, hi, 0,1,2,3,4,5,6,7,8,9,10,11,12,13,14,15);
        }
        // B 32x16 bf16 frag: lane<16 -> n=lane,    K 0..7 then 16..23
        //                    lane>=16 -> n=lane-16, K 8..15 then 24..31
        v16bf bfrag[2];
#pragma unroll
        for (int tj = 0; tj < 2; ++tj) {
            int n  = wn + tj * 16 + (lane & 15);
            int kb = (lane < 16) ? 0 : 8;
            v8bf lo = *(const v8bf*)&sB[buf][n][kb];
            v8bf hi = *(const v8bf*)&sB[buf][n][kb + 16];
            bfrag[tj] = __builtin_shufflevector(lo, hi, 0,1,2,3,4,5,6,7,8,9,10,11,12,13,14,15);
        }

#pragma unroll
        for (int ti = 0; ti < 2; ++ti)
#pragma unroll
            for (int tj = 0; tj < 2; ++tj)
                acc[ti][tj] = __builtin_amdgcn_wmma_f32_16x16x32_bf16(
                    false, afrag[ti], false, bfrag[tj],
                    (short)0, acc[ti][tj], false, false);

        __syncthreads();   // everyone done reading buf before it is refilled
    }

    // Epilogue. C/D layout: elem r of v8f -> M = r + (lane>=16 ? 8:0), N = lane&15
    int nobj = 0;
    if (EPI == EPI_ADJ) nobj = num_obj[bz];
    const int col   = lane & 15;
    const int rbase = (lane >> 4) * 8;
#pragma unroll
    for (int ti = 0; ti < 2; ++ti)
#pragma unroll
        for (int tj = 0; tj < 2; ++tj)
#pragma unroll
            for (int r = 0; r < 8; ++r) {
                int grow = gm + wm + ti * 16 + rbase + r;
                int gcol = gn + wn + tj * 16 + col;
                float v  = acc[ti][tj][r];
                if (bias) v += bias[gcol];
                float o;
                if (EPI == EPI_BF16)            o = v;
                else if (EPI == EPI_RELU_BF16)  o = (v > 0.f) ? v : 0.f;
                else { // adjacency: sigmoid(v)>0.5 <=> v>0, masked by valid nodes
                    o = ((v > 0.f) && (grow < nobj) && (gcol < nobj)) ? 1.f : 0.f;
                }
                __bf16 ob = (__bf16)o;
                Cg[(size_t)bz * sCb + (size_t)grow * N + gcol] = ob;
                if (CgT) { // per-128-row-group transposed copy (GNN h^T)
                    int bb = grow >> 7, nn = grow & 127;
                    CgT[((size_t)bb * N << 7) + ((size_t)gcol << 7) + nn] = ob;
                }
            }
}

// ---------------------------------------------------------------------------
// Elementwise helpers
// ---------------------------------------------------------------------------
__global__ void k_f32_to_bf16(const float* __restrict__ in, __bf16* __restrict__ out,
                              long long n)
{
    long long i = (long long)blockIdx.x * blockDim.x + threadIdx.x;
    long long s = (long long)gridDim.x * blockDim.x;
    for (; i < n; i += s) out[i] = (__bf16)in[i];
}

// Batched transpose + convert: in [b][R][C] f32 -> out [b][C][R] bf16
__global__ void k_cvt_T(const float* __restrict__ in, __bf16* __restrict__ out,
                        int R, int C, long long total)
{
    long long i = (long long)blockIdx.x * blockDim.x + threadIdx.x;
    long long s = (long long)gridDim.x * blockDim.x;
    const long long rc = (long long)R * C;
    for (; i < total; i += s) {
        long long b = i / rc, rem = i - b * rc;
        int c = (int)(rem / R), r = (int)(rem % R);
        out[i] = (__bf16)in[b * rc + (long long)r * C + c];
    }
}

// OR forced (ground-truth) edges into the adjacency, symmetric.
__global__ void k_force_edges(const int* __restrict__ pairs, const int* __restrict__ num_edges,
                              __bf16* __restrict__ adj, int Bn, int Pn, int Nn)
{
    int idx = blockIdx.x * blockDim.x + threadIdx.x;
    if (idx >= Bn * Pn) return;
    int b = idx / Pn, p = idx - b * Pn;
    if (p >= num_edges[b]) return;
    int i0 = pairs[idx * 2 + 0];
    int i1 = pairs[idx * 2 + 1];
    adj[(size_t)b * Nn * Nn + (size_t)i0 * Nn + i1] = (__bf16)1.f;
    adj[(size_t)b * Nn * Nn + (size_t)i1 * Nn + i0] = (__bf16)1.f;
}

// cls_in[bp, d] = 0.5*(h[b, i0, d] + h[b, i1, d])
__global__ void k_gather_pairs(const __bf16* __restrict__ h, const int* __restrict__ pairs,
                               __bf16* __restrict__ cls, int Bn, int Pn, int Nn, int Dn)
{
    long long idx   = (long long)blockIdx.x * blockDim.x + threadIdx.x;
    long long total = (long long)Bn * Pn * Dn;
    if (idx >= total) return;
    int d  = (int)(idx % Dn);
    long long bp = idx / Dn;
    int b  = (int)(bp / Pn);
    int i0 = pairs[bp * 2 + 0];
    int i1 = pairs[bp * 2 + 1];
    float v = 0.5f * ((float)h[((size_t)b * Nn + i0) * Dn + d] +
                      (float)h[((size_t)b * Nn + i1) * Dn + d]);
    cls[idx] = (__bf16)v;
}

// Final tiny layer per head: out[bp, j] = hid[bp,:] @ w1[:, j] + b1[j]  (opt. softmax)
__global__ void k_head_final(const __bf16* __restrict__ hid, const float* __restrict__ w1,
                             const float* __restrict__ b1, float* __restrict__ out,
                             int BP, int KK, int OUT, int do_softmax)
{
    int bp = blockIdx.x * blockDim.x + threadIdx.x;
    if (bp >= BP) return;
    float z[6];
    for (int j = 0; j < OUT; ++j) z[j] = b1[j];
    const __bf16* hr = hid + (size_t)bp * KK;
    for (int k = 0; k < KK; ++k) {
        float hv = (float)hr[k];
        for (int j = 0; j < OUT; ++j) z[j] += hv * w1[k * OUT + j];
    }
    if (do_softmax) {
        float m = z[0];
        for (int j = 1; j < OUT; ++j) m = fmaxf(m, z[j]);
        float s = 0.f;
        for (int j = 0; j < OUT; ++j) { z[j] = __expf(z[j] - m); s += z[j]; }
        float inv = 1.f / s;
        for (int j = 0; j < OUT; ++j) z[j] *= inv;
    }
    for (int j = 0; j < OUT; ++j) out[(size_t)bp * OUT + j] = z[j];
}

// ---------------------------------------------------------------------------
// Host orchestration
// ---------------------------------------------------------------------------
extern "C" void kernel_launch(void* const* d_in, const int* in_sizes, int n_in,
                              void* d_out, int out_size, void* d_ws, size_t ws_size,
                              hipStream_t stream)
{
    (void)in_sizes; (void)n_in; (void)out_size; (void)ws_size;
    const int  Bsz = 256, Nn = 128, Dd = 512, Pp = 64;
    const long long MN = (long long)Bsz * Nn;         // 32768 nodes total
    const int  BP  = Bsz * Pp;                        // 16384 pairs total

    // ---- inputs (setup_inputs order) ----
    const float* feats      = (const float*)d_in[0];
    const int*   num_obj    = (const int*)d_in[1];
    const int*   num_edges  = (const int*)d_in[2];
    const int*   pairs      = (const int*)d_in[3];
    const float* phi_w0 = (const float*)d_in[4];  const float* phi_b0 = (const float*)d_in[5];
    const float* phi_w1 = (const float*)d_in[6];  const float* phi_b1 = (const float*)d_in[7];
    const float* psi_w0 = (const float*)d_in[8];  const float* psi_b0 = (const float*)d_in[9];
    const float* psi_w1 = (const float*)d_in[10]; const float* psi_b1 = (const float*)d_in[11];
    const float* gnn_w0 = (const float*)d_in[12]; const float* gnn_b0 = (const float*)d_in[13];
    const float* gnn_w1 = (const float*)d_in[14]; const float* gnn_b1 = (const float*)d_in[15];
    const float* cr_w0  = (const float*)d_in[16]; const float* cr_b0  = (const float*)d_in[17];
    const float* cr_w1  = (const float*)d_in[18]; const float* cr_b1  = (const float*)d_in[19];
    const float* lr_w0  = (const float*)d_in[20]; const float* lr_b0  = (const float*)d_in[21];
    const float* lr_w1  = (const float*)d_in[22]; const float* lr_b1  = (const float*)d_in[23];
    const float* mr_w0  = (const float*)d_in[24]; const float* mr_b0  = (const float*)d_in[25];
    const float* mr_w1  = (const float*)d_in[26]; const float* mr_b1  = (const float*)d_in[27];

    // ---- workspace carve-up ----
    size_t off = 0;
    auto carve = [&](size_t bytes) {
        off = (off + 255) & ~(size_t)255;
        size_t o = off; off += bytes;
        return (char*)d_ws + o;
    };
    __bf16* feats_bf = (__bf16*)carve(MN * Dd * 2);            // feats bf16; reused as h1/h2
    __bf16* featsT   = (__bf16*)carve(MN * Dd * 2);            // per-batch [D][N] transpose
    __bf16* hT       = (__bf16*)carve(MN * Dd * 2);            // h1 transposed, per-batch [D][N]
    __bf16* hid256   = (__bf16*)carve(MN * 256 * 2);           // MLP hidden (reused)
    __bf16* phi_b    = (__bf16*)carve(MN * 64 * 2);
    __bf16* psi_b    = (__bf16*)carve(MN * 64 * 2);
    __bf16* adjb     = (__bf16*)carve((size_t)Bsz * Nn * Nn * 2);
    __bf16* msgb     = (__bf16*)carve(MN * Dd * 2);
    __bf16* clsb     = (__bf16*)carve((size_t)BP * Dd * 2);
    // all weights stored transposed: wXT is [N,K] bf16
    __bf16* wphi0T = (__bf16*)carve(512 * 256 * 2);
    __bf16* wphi1T = (__bf16*)carve(256 * 64 * 2);
    __bf16* wpsi0T = (__bf16*)carve(512 * 256 * 2);
    __bf16* wpsi1T = (__bf16*)carve(256 * 64 * 2);
    __bf16* wg0T   = (__bf16*)carve(512 * 512 * 2);
    __bf16* wg1T   = (__bf16*)carve(512 * 512 * 2);
    __bf16* wcr0T  = (__bf16*)carve(512 * 256 * 2);
    __bf16* wlr0T  = (__bf16*)carve(512 * 256 * 2);
    __bf16* wmr0T  = (__bf16*)carve(512 * 256 * 2);

    auto blocks_for = [](long long n) {
        long long b = (n + 255) / 256;
        return (int)(b > 4096 ? 4096 : b);
    };
    auto cvt = [&](const float* src, __bf16* dst, long long n) {
        k_f32_to_bf16<<<blocks_for(n), 256, 0, stream>>>(src, dst, n);
    };
    auto cvtT = [&](const float* src, __bf16* dst, int R, int C, long long total) {
        k_cvt_T<<<blocks_for(total), 256, 0, stream>>>(src, dst, R, C, total);
    };

    // ---- conversions (weights transposed to [N,K]) ----
    cvt (feats,  feats_bf, MN * Dd);
    cvtT(feats,  featsT, Nn, Dd, MN * Dd);          // [B][128][512] -> [B][512][128]
    cvtT(phi_w0, wphi0T, 512, 256, 512 * 256);  cvtT(phi_w1, wphi1T, 256, 64, 256 * 64);
    cvtT(psi_w0, wpsi0T, 512, 256, 512 * 256);  cvtT(psi_w1, wpsi1T, 256, 64, 256 * 64);
    cvtT(gnn_w0, wg0T, 512, 512, 512 * 512);    cvtT(gnn_w1, wg1T, 512, 512, 512 * 512);
    cvtT(cr_w0,  wcr0T, 512, 256, 512 * 256);
    cvtT(lr_w0,  wlr0T, 512, 256, 512 * 256);
    cvtT(mr_w0,  wmr0T, 512, 256, 512 * 256);

    const dim3 blk(128);
    const int MT = (int)(MN / 64);     // 512 row-tiles for node-level GEMMs
    const int PT = BP / 64;            // 256 row-tiles for pair-level GEMMs

    // ---- relation proposal: phi / psi MLPs ----
    gemm_wmma_bf16<EPI_RELU_BF16><<<dim3(4, MT, 1), blk, 0, stream>>>(
        feats_bf, wphi0T, phi_b0, hid256, nullptr, (int)MN, 256, 512, 0, 0, 0, nullptr);
    gemm_wmma_bf16<EPI_BF16><<<dim3(1, MT, 1), blk, 0, stream>>>(
        hid256, wphi1T, phi_b1, phi_b, nullptr, (int)MN, 64, 256, 0, 0, 0, nullptr);
    gemm_wmma_bf16<EPI_RELU_BF16><<<dim3(4, MT, 1), blk, 0, stream>>>(
        feats_bf, wpsi0T, psi_b0, hid256, nullptr, (int)MN, 256, 512, 0, 0, 0, nullptr);
    gemm_wmma_bf16<EPI_BF16><<<dim3(1, MT, 1), blk, 0, stream>>>(
        hid256, wpsi1T, psi_b1, psi_b, nullptr, (int)MN, 64, 256, 0, 0, 0, nullptr);

    // ---- adjacency: batched phi @ psi^T (psi already [N,K]) ----
    gemm_wmma_bf16<EPI_ADJ><<<dim3(2, 2, Bsz), blk, 0, stream>>>(
        phi_b, psi_b, nullptr, adjb, nullptr, Nn, Nn, 64,
        (long long)Nn * 64, (long long)Nn * 64, (long long)Nn * Nn, num_obj);
    k_force_edges<<<(BP + 255) / 256, 256, 0, stream>>>(pairs, num_edges, adjb, Bsz, Pp, Nn);

    // ---- GNN layer 0: msg0 = Adj @ feats ; h1 = relu(msg0 @ W0 + b0) (+h1^T) ----
    gemm_wmma_bf16<EPI_BF16><<<dim3(8, 2, Bsz), blk, 0, stream>>>(
        adjb, featsT, nullptr, msgb, nullptr, Nn, Dd, Nn,
        (long long)Nn * Nn, (long long)Nn * Dd, (long long)Nn * Dd, nullptr);
    gemm_wmma_bf16<EPI_RELU_BF16><<<dim3(8, MT, 1), blk, 0, stream>>>(
        msgb, wg0T, gnn_b0, feats_bf, hT, (int)MN, 512, 512, 0, 0, 0, nullptr);

    // ---- GNN layer 1: msg1 = Adj @ h1 ; h2 = relu(msg1 @ W1 + b1) ----
    gemm_wmma_bf16<EPI_BF16><<<dim3(8, 2, Bsz), blk, 0, stream>>>(
        adjb, hT, nullptr, msgb, nullptr, Nn, Dd, Nn,
        (long long)Nn * Nn, (long long)Nn * Dd, (long long)Nn * Dd, nullptr);
    gemm_wmma_bf16<EPI_RELU_BF16><<<dim3(8, MT, 1), blk, 0, stream>>>(
        msgb, wg1T, gnn_b1, feats_bf, nullptr, (int)MN, 512, 512, 0, 0, 0, nullptr);

    // ---- pair gather: cls_in = 0.5*(h[i0] + h[i1]) ----
    {
        long long total = (long long)BP * Dd;
        k_gather_pairs<<<(int)((total + 255) / 256), 256, 0, stream>>>(
            feats_bf, pairs, clsb, Bsz, Pp, Nn, Dd);
    }

    // ---- heads (output order: lr, cr, mr) ----
    float* out_lr = (float*)d_out;
    float* out_cr = out_lr + (size_t)BP * 4;
    float* out_mr = out_cr + (size_t)BP * 5;

    gemm_wmma_bf16<EPI_RELU_BF16><<<dim3(4, PT, 1), blk, 0, stream>>>(
        clsb, wlr0T, lr_b0, hid256, nullptr, BP, 256, 512, 0, 0, 0, nullptr);
    k_head_final<<<(BP + 255) / 256, 256, 0, stream>>>(hid256, lr_w1, lr_b1, out_lr, BP, 256, 4, 0);

    gemm_wmma_bf16<EPI_RELU_BF16><<<dim3(4, PT, 1), blk, 0, stream>>>(
        clsb, wcr0T, cr_b0, hid256, nullptr, BP, 256, 512, 0, 0, 0, nullptr);
    k_head_final<<<(BP + 255) / 256, 256, 0, stream>>>(hid256, cr_w1, cr_b1, out_cr, BP, 256, 5, 1);

    gemm_wmma_bf16<EPI_RELU_BF16><<<dim3(4, PT, 1), blk, 0, stream>>>(
        clsb, wmr0T, mr_b0, hid256, nullptr, BP, 256, 512, 0, 0, 0, nullptr);
    k_head_final<<<(BP + 255) / 256, 256, 0, stream>>>(hid256, mr_w1, mr_b1, out_mr, BP, 256, 3, 0);
}
